// BalancedTopkMLP_52003464020213
// MI455X (gfx1250) — compile-verified, matrix operand order checked by reading, not run on hardware
//
#include <hip/hip_runtime.h>
#include <hip/hip_bf16.h>

// BalancedTopkMLP for MI455X (gfx1250, wave32, WMMA).
// HBM-bound (~2.3 GB traffic @ 23.3 TB/s >> 2.47 TFLOP bf16 WMMA).
// Dense bf16 WMMA GEMMs; fp32->bf16 conversion during global->LDS staging
// via v_add + v_perm_b32 (2 elems / 3 VALU).  The gate/up epilogue stores h
// in bf16 so the down-GEMM streams a 180 MB (not 360 MB) activation buffer
// and needs no A-side conversion at all.
// Grid sweep order per GEMM keeps the L2(192MB)-fitting operand resident:
//   gate/up: W=360MB>L2, A=x=128MB<L2     -> M-tiles fast
//   down:    A=h=180MB,  W=180MB<L2       -> N-tiles fast (W resident)
//   pred1/2: both fit                     -> M-tiles fast

typedef __attribute__((ext_vector_type(16))) __bf16 v16bf;
typedef __attribute__((ext_vector_type(8)))  float  v8f;

constexpr int HD     = 4096;          // hidden
constexpr int ID     = 11008;         // intermediate (= 86 * 128)
constexpr int PD     = 1024;          // predictor hidden
constexpr int BANKSZ = 128;
constexpr int KSEL   = 16;
constexpr int NBANK  = ID / BANKSZ;   // 86
constexpr int NTOK   = 4 * 2048;      // 8192 tokens

// Pack two fp32 -> two bf16 (round-to-nearest, ties away): 2 adds + 1 perm.
static __device__ __forceinline__ unsigned pack2bf(unsigned u0, unsigned u1) {
    u0 += 0x8000u; u1 += 0x8000u;
#if __has_builtin(__builtin_amdgcn_perm)
    // out = {u1[31:16], u0[31:16]}; combined qword = {src0,src1}, sel 0-3 = src1
    return __builtin_amdgcn_perm(u1, u0, 0x07060302u);
#else
    return (u0 >> 16) | (u1 & 0xFFFF0000u);
#endif
}

static __device__ __forceinline__ __bf16 f2bf(float f) {
    union { float f; unsigned u; } a; a.f = f;
    union { unsigned short s; __bf16 b; } o;
    o.s = (unsigned short)((a.u + 0x8000u) >> 16);
    return o.b;
}

// 16 fp32 (4 float4) -> 16 bf16 packed as two 16-byte chunks.
union BFPack { __bf16 h[16]; unsigned w[8]; uint4 q[2]; };

static __device__ __forceinline__ void cvt16(const float4* r, BFPack& pk) {
    const unsigned* f = (const unsigned*)r;
#pragma unroll
    for (int c = 0; c < 8; ++c) pk.w[c] = pack2bf(f[2 * c], f[2 * c + 1]);
}

// Build a v16bf WMMA fragment from two 16-byte LDS chunks
// (K in [kb,kb+8) and [kb+16,kb+24) per the 16-bit operand layout).
static __device__ __forceinline__ v16bf ld_frag(const __bf16* lo, const __bf16* hi) {
    union { v16bf v; uint4 q[2]; } u;
    u.q[0] = *(const uint4*)lo;
    u.q[1] = *(const uint4*)hi;
    return u.v;
}

static __device__ __forceinline__ float fsigmoid(float x) {
    return 1.0f / (1.0f + __expf(-x));
}

// -------------------------------------------------------------------------
// C[M,N] = A[M,K] @ W[N,K]^T.  EPI: 0 = none, 1 = sigmoid.
// MFAST: 1 -> M tiles on blockIdx.x (A stays L2-resident),
//        0 -> N tiles on blockIdx.x (W stays L2-resident).
// ABF:   1 -> A is already bf16 (direct 32B copy to LDS, no conversion).
// Block tile 128x128, BK=32, 8 waves (2x4), wave tile 64x32 (4x2 WMMA tiles).
// -------------------------------------------------------------------------
template <int EPI, int MFAST, int ABF>
__global__ __launch_bounds__(256) void gemm_xwT(
    const void* __restrict__ Av, const float* __restrict__ W,
    float* __restrict__ C, int M, int N, int K)
{
    __shared__ __bf16 sA[128][40];  // stride 40 (80B) to spread LDS banks
    __shared__ __bf16 sB[128][40];

    const int tid = threadIdx.x;
    const int wv  = tid >> 5;
    const int ln  = tid & 31;
    const int wm  = wv >> 2;                 // 0..1
    const int wn  = wv & 3;                  // 0..3
    const long tileM = (long)(MFAST ? blockIdx.x : blockIdx.y) * 128;
    const long tileN = (long)(MFAST ? blockIdx.y : blockIdx.x) * 128;

    // Global staging: thread t -> row t>>1, K-offset (t&1)*16 (16 elements).
    const int  lrow  = tid >> 1;
    const int  lkoff = (tid & 1) * 16;
    const long aoff  = (tileM + lrow) * (long)K + lkoff;
    const float*  gAf = (const float*)Av + aoff;
    const __bf16* gAh = (const __bf16*)Av + aoff;
    const float*  gB  = W + (tileN + lrow) * (long)K + lkoff;

    v8f acc[4][2];
    const v8f vzero = {0.f,0.f,0.f,0.f,0.f,0.f,0.f,0.f};
#pragma unroll
    for (int mt = 0; mt < 4; ++mt)
#pragma unroll
        for (int nt = 0; nt < 2; ++nt) acc[mt][nt] = vzero;

    float4 ra[4], rb[4];
    uint4  rah[2];
    if constexpr (ABF) {
        rah[0] = *(const uint4*)(gAh);
        rah[1] = *(const uint4*)(gAh + 8);
    } else {
#pragma unroll
        for (int i = 0; i < 4; ++i) ra[i] = *(const float4*)(gAf + i * 4);
    }
#pragma unroll
    for (int i = 0; i < 4; ++i) rb[i] = *(const float4*)(gB + i * 4);
    {
        if constexpr (ABF) {
            *(uint4*)&sA[lrow][lkoff]     = rah[0];
            *(uint4*)&sA[lrow][lkoff + 8] = rah[1];
        } else {
            BFPack pa; cvt16(ra, pa);
            *(uint4*)&sA[lrow][lkoff]     = pa.q[0];
            *(uint4*)&sA[lrow][lkoff + 8] = pa.q[1];
        }
        BFPack pb; cvt16(rb, pb);
        *(uint4*)&sB[lrow][lkoff]     = pb.q[0];
        *(uint4*)&sB[lrow][lkoff + 8] = pb.q[1];
    }
    __syncthreads();

    const int KT = K >> 5;
    const int kb = (ln >> 4) << 3;           // 0 or 8
    for (int kt = 0; kt < KT; ++kt) {
        if (kt + 1 < KT) {                   // register prefetch of next K-slab
            const long ks = (long)(kt + 1) * 32;
            if constexpr (ABF) {
                rah[0] = *(const uint4*)(gAh + ks);
                rah[1] = *(const uint4*)(gAh + ks + 8);
            } else {
                const float* pA = gAf + ks;
#pragma unroll
                for (int i = 0; i < 4; ++i) ra[i] = *(const float4*)(pA + i * 4);
            }
            const float* pB = gB + ks;
#pragma unroll
            for (int i = 0; i < 4; ++i) rb[i] = *(const float4*)(pB + i * 4);
        }
        v16bf af[4], bfr[2];
#pragma unroll
        for (int mt = 0; mt < 4; ++mt) {
            const __bf16* r = sA[wm * 64 + mt * 16 + (ln & 15)];
            af[mt] = ld_frag(r + kb, r + kb + 16);
        }
#pragma unroll
        for (int nt = 0; nt < 2; ++nt) {
            const __bf16* r = sB[wn * 32 + nt * 16 + (ln & 15)];
            bfr[nt] = ld_frag(r + kb, r + kb + 16);
        }
#pragma unroll
        for (int mt = 0; mt < 4; ++mt)
#pragma unroll
            for (int nt = 0; nt < 2; ++nt)
                acc[mt][nt] = __builtin_amdgcn_wmma_f32_16x16x32_bf16(
                    false, af[mt], false, bfr[nt], (short)0, acc[mt][nt], false, false);
        __syncthreads();
        if (kt + 1 < KT) {
            if constexpr (ABF) {
                *(uint4*)&sA[lrow][lkoff]     = rah[0];
                *(uint4*)&sA[lrow][lkoff + 8] = rah[1];
            } else {
                BFPack pa; cvt16(ra, pa);
                *(uint4*)&sA[lrow][lkoff]     = pa.q[0];
                *(uint4*)&sA[lrow][lkoff + 8] = pa.q[1];
            }
            BFPack pb; cvt16(rb, pb);
            *(uint4*)&sB[lrow][lkoff]     = pb.q[0];
            *(uint4*)&sB[lrow][lkoff + 8] = pb.q[1];
            __syncthreads();
        }
    }

    // C/D layout: VGPR v -> row v (+8 for lanes 16-31), col = lane&15.
    const int rbase = wm * 64 + ((ln >> 4) << 3);
    const long cbase = tileN + wn * 32 + (ln & 15);
#pragma unroll
    for (int mt = 0; mt < 4; ++mt)
#pragma unroll
        for (int nt = 0; nt < 2; ++nt)
#pragma unroll
            for (int v = 0; v < 8; ++v) {
                long row = tileM + rbase + mt * 16 + v;
                float val = acc[mt][nt][v];
                if (EPI == 1) val = fsigmoid(val);
                C[row * (long)N + cbase + nt * 16] = val;
            }
}

// -------------------------------------------------------------------------
// Per-(token,bank) balanced top-16 of 128 on |pred|+bias; in-place pred*mask.
// One wave32 per (token,bank); 4 candidates per lane; 16 shuffle-argmax rounds.
// -------------------------------------------------------------------------
__global__ __launch_bounds__(256) void balanced_topk(
    float* __restrict__ PR, const float* __restrict__ bias)
{
    const int wv = threadIdx.x >> 5;
    const int ln = threadIdx.x & 31;
    const long gid = (long)blockIdx.x * 8 + wv;   // token * NBANK + bank
    const long token = gid / NBANK;
    const int  bank  = (int)(gid % NBANK);
    float* p = PR + token * (long)ID + (long)bank * BANKSZ;
    const float* bb = bias + bank * BANKSZ;

    float pv[4], sc[4]; int keep[4];
#pragma unroll
    for (int j = 0; j < 4; ++j) {
        int i = ln + 32 * j;
        pv[j] = p[i];
        sc[j] = fabsf(pv[j]) + bb[i];
        keep[j] = 0;
    }
#pragma unroll
    for (int r = 0; r < KSEL; ++r) {
        float bv = sc[0]; int bj = 0;
#pragma unroll
        for (int j = 1; j < 4; ++j) if (sc[j] > bv) { bv = sc[j]; bj = j; }
        float v = bv; int s = ln;
#pragma unroll
        for (int off = 16; off > 0; off >>= 1) {
            float ov = __shfl_xor(v, off, 32);
            int   os = __shfl_xor(s, off, 32);
            if (ov > v || (ov == v && os < s)) { v = ov; s = os; }
        }
        if (s == ln) { keep[bj] = 1; sc[bj] = -3.4e38f; }
    }
#pragma unroll
    for (int j = 0; j < 4; ++j) p[ln + 32 * j] = keep[j] ? pv[j] : 0.0f;
}

// -------------------------------------------------------------------------
// Fused gate/up GEMMs + epilogue: HOUT = bf16(MP * silu(x@Wg^T) * (x@Wu^T)).
// Block tile 128x64, 8 waves (4x2), wave tile 32x32; two accumulator sets.
// M tiles on blockIdx.x so x (128 MB) stays L2-resident while the 360 MB of
// gate/up weights stream from HBM exactly once.
// -------------------------------------------------------------------------
__global__ __launch_bounds__(256) void gemm_gateup(
    const float* __restrict__ X, const float* __restrict__ WG,
    const float* __restrict__ WU, const float* __restrict__ MP,
    __bf16* __restrict__ HOUT)
{
    __shared__ __bf16 sA[128][40];
    __shared__ __bf16 sG[64][40];
    __shared__ __bf16 sU[64][40];

    const int tid = threadIdx.x;
    const int wv  = tid >> 5;
    const int ln  = tid & 31;
    const int wm  = wv >> 1;                 // 0..3
    const int wn  = wv & 1;                  // 0..1
    const long tileM = (long)blockIdx.x * 128;   // M fast
    const long tileN = (long)blockIdx.y * 64;
    const int K = HD;

    // A staging: thread -> row tid>>1, koff (tid&1)*16.
    // W staging: threads 0..127 -> WG rows, 128..255 -> WU rows.
    const int  arow  = tid >> 1;
    const int  akoff = (tid & 1) * 16;
    const int  wrow  = (tid & 127) >> 1;
    const float* gA = X + (tileM + arow) * (long)K + akoff;
    const float* gW = (tid < 128 ? WG : WU) + (tileN + wrow) * (long)K + akoff;
    __bf16* sW = (tid < 128 ? &sG[0][0] : &sU[0][0]) + wrow * 40 + akoff;

    v8f accg[2][2], accu[2][2];
    const v8f vzero = {0.f,0.f,0.f,0.f,0.f,0.f,0.f,0.f};
#pragma unroll
    for (int mt = 0; mt < 2; ++mt)
#pragma unroll
        for (int nt = 0; nt < 2; ++nt) { accg[mt][nt] = vzero; accu[mt][nt] = vzero; }

    float4 ra[4], rw[4];
#pragma unroll
    for (int i = 0; i < 4; ++i) { ra[i] = *(const float4*)(gA + i * 4); rw[i] = *(const float4*)(gW + i * 4); }
    {
        BFPack pa, pw; cvt16(ra, pa); cvt16(rw, pw);
        *(uint4*)&sA[arow][akoff]     = pa.q[0];
        *(uint4*)&sA[arow][akoff + 8] = pa.q[1];
        *(uint4*)(sW)                 = pw.q[0];
        *(uint4*)(sW + 8)             = pw.q[1];
    }
    __syncthreads();

    const int KT = K >> 5;
    const int kb = (ln >> 4) << 3;
    for (int kt = 0; kt < KT; ++kt) {
        if (kt + 1 < KT) {
            const float* pA = gA + (long)(kt + 1) * 32;
            const float* pW = gW + (long)(kt + 1) * 32;
#pragma unroll
            for (int i = 0; i < 4; ++i) { ra[i] = *(const float4*)(pA + i * 4); rw[i] = *(const float4*)(pW + i * 4); }
        }
        v16bf af[2], bg[2], bu[2];
#pragma unroll
        for (int mt = 0; mt < 2; ++mt) {
            const __bf16* r = sA[wm * 32 + mt * 16 + (ln & 15)];
            af[mt] = ld_frag(r + kb, r + kb + 16);
        }
#pragma unroll
        for (int nt = 0; nt < 2; ++nt) {
            const __bf16* rg = sG[wn * 32 + nt * 16 + (ln & 15)];
            const __bf16* ru = sU[wn * 32 + nt * 16 + (ln & 15)];
            bg[nt] = ld_frag(rg + kb, rg + kb + 16);
            bu[nt] = ld_frag(ru + kb, ru + kb + 16);
        }
#pragma unroll
        for (int mt = 0; mt < 2; ++mt)
#pragma unroll
            for (int nt = 0; nt < 2; ++nt) {
                accg[mt][nt] = __builtin_amdgcn_wmma_f32_16x16x32_bf16(
                    false, af[mt], false, bg[nt], (short)0, accg[mt][nt], false, false);
                accu[mt][nt] = __builtin_amdgcn_wmma_f32_16x16x32_bf16(
                    false, af[mt], false, bu[nt], (short)0, accu[mt][nt], false, false);
            }
        __syncthreads();
        if (kt + 1 < KT) {
            BFPack pa, pw; cvt16(ra, pa); cvt16(rw, pw);
            *(uint4*)&sA[arow][akoff]     = pa.q[0];
            *(uint4*)&sA[arow][akoff + 8] = pa.q[1];
            *(uint4*)(sW)                 = pw.q[0];
            *(uint4*)(sW + 8)             = pw.q[1];
            __syncthreads();
        }
    }

    const int rbase = wm * 32 + ((ln >> 4) << 3);
    const long cbase = tileN + wn * 32 + (ln & 15);
#pragma unroll
    for (int mt = 0; mt < 2; ++mt)
#pragma unroll
        for (int nt = 0; nt < 2; ++nt)
#pragma unroll
            for (int v = 0; v < 8; ++v) {
                long row = tileM + rbase + mt * 16 + v;
                long idx = row * (long)ID + cbase + nt * 16;
                float g = accg[mt][nt][v];
                float u = accu[mt][nt][v];
                float mp = MP[idx];                       // masked pred (0 if inactive)
                HOUT[idx] = f2bf(mp * g * fsigmoid(g) * u);
            }
}

// -------------------------------------------------------------------------
extern "C" void kernel_launch(void* const* d_in, const int* in_sizes, int n_in,
                              void* d_out, int out_size, void* d_ws, size_t ws_size,
                              hipStream_t stream)
{
    const float* x       = (const float*)d_in[0];  // [NTOK, HD]
    const float* w_pred1 = (const float*)d_in[1];  // [PD, HD]
    const float* w_pred2 = (const float*)d_in[2];  // [ID, PD]
    const float* w_gate  = (const float*)d_in[3];  // [ID, HD]
    const float* w_up    = (const float*)d_in[4];  // [ID, HD]
    const float* w_down  = (const float*)d_in[5];  // [HD, ID]
    const float* bias    = (const float*)d_in[6];  // [ID]
    float* out = (float*)d_out;                    // [NTOK, HD]

    // Workspace: buf0 fp32 [NTOK,PD] (32 MB) + buf1 fp32 [NTOK,ID] (360.5 MB)
    //            + buf2 bf16 [NTOK,ID] (180 MB)  ~= 574 MB total.
    float*  buf0 = (float*)d_ws;                        // predictor hidden
    float*  buf1 = buf0 + (size_t)NTOK * PD;            // pred -> masked pred
    __bf16* buf2 = (__bf16*)(buf1 + (size_t)NTOK * ID); // h (bf16)

    dim3 blk(256);
    // 1) pred hidden: [NTOK,PD] = x @ w_pred1^T          (M fast; x L2-resident)
    gemm_xwT<0, 1, 0><<<dim3(NTOK / 128, PD / 128), blk, 0, stream>>>(x, w_pred1, buf0, NTOK, PD, HD);
    // 2) pred = sigmoid(buf0 @ w_pred2^T) : [NTOK,ID]    (M fast; buf0 L2-resident)
    gemm_xwT<1, 1, 0><<<dim3(NTOK / 128, ID / 128), blk, 0, stream>>>(buf0, w_pred2, buf1, NTOK, ID, PD);
    // 3) per-bank top-16 mask, in place
    balanced_topk<<<dim3((NTOK * NBANK) / 8), blk, 0, stream>>>(buf1, bias);
    // 4) h(bf16) = maskedpred * silu(x@Wg^T) * (x@Wu^T)  (M fast)
    gemm_gateup<<<dim3(NTOK / 128, ID / 64), blk, 0, stream>>>(x, w_gate, w_up, buf1, buf2);
    // 5) out = h @ w_down^T, bf16 A path                 (N fast; w_down L2-resident)
    gemm_xwT<0, 0, 1><<<dim3(HD / 128, NTOK / 128), blk, 0, stream>>>(buf2, w_down, out, NTOK, HD, ID);
}
